// EncoderBlock_43499428774108
// MI455X (gfx1250) — compile-verified, matrix operand order checked by reading, not run on hardware
//
#include <hip/hip_runtime.h>
#include <hip/hip_bf16.h>
#include <cstdint>

// MI455X / gfx1250 transformer encoder block.
// All matmuls (QKV proj, QK^T, PV, O-proj, FFN1, FFN2) run on
// v_wmma_f32_16x16x32_f16 (f16 in, f32 accumulate). Tile staging uses
// GLOBAL_LOAD_ASYNC_TO_LDS_B128 (ASYNCcnt) when the builtin is available,
// with double-buffered LDS tiles in the GEMM for copy/compute overlap.

typedef _Float16 half_t;
typedef _Float16 v16h __attribute__((ext_vector_type(16)));
typedef _Float16 v8h  __attribute__((ext_vector_type(8)));
typedef float    v8f  __attribute__((ext_vector_type(8)));
typedef int      v4i  __attribute__((vector_size(16)));   // builtin pointee type

#define D_MODEL 1024
#define HEADS   16
#define D_K     64
#define D_FF    4096
#define SEQ     1024
#define BATCH   8
#define M_TOT   (BATCH * SEQ)   // 8192 rows
#define LN_EPS  1e-6f

#if defined(__has_builtin)
#if __has_builtin(__builtin_amdgcn_global_load_async_to_lds_b128) && \
    __has_builtin(__builtin_amdgcn_s_wait_asynccnt)
#define USE_ASYNC_LDS 1
#endif
#endif
#ifndef USE_ASYNC_LDS
#define USE_ASYNC_LDS 0
#endif

union HV { v16h v; v8h h[2]; };

__device__ inline v8f wmma16(v16h a, v16h b, v8f c) {
  // D = A(16x32 f16) * B(32x16 f16) + C(16x16 f32)
  return __builtin_amdgcn_wmma_f32_16x16x32_f16(false, a, false, b, (short)0, c,
                                                false, false);
}

// Copy 16 bytes global -> LDS. Async (ASYNCcnt) on CDNA5 when available.
__device__ inline void cp16_g2l(const half_t* g, half_t* l) {
#if USE_ASYNC_LDS
  // Flat LDS address truncates to the 32-bit LDS offset (ISA 10.2).
  __builtin_amdgcn_global_load_async_to_lds_b128(
      (__attribute__((address_space(1))) v4i*)(uintptr_t)g,
      (__attribute__((address_space(3))) v4i*)(uint32_t)(uintptr_t)l, 0, 0);
#else
  *(v8h*)l = *(const v8h*)g;
#endif
}

__device__ inline void async_fence() {
#if USE_ASYNC_LDS
  __builtin_amdgcn_s_wait_asynccnt(0);
#endif
}

// Load one lane's 16 f16 fragment values from a row-major [row][K] LDS tile.
// k-values: {sel..sel+7, 16+sel..16+sel+7}, sel = 0 (lanes 0-15) or 8 (16-31).
__device__ inline v16h frag_ld(const half_t* rowbase, int sel) {
  HV u;
  u.h[0] = *(const v8h*)(rowbase + sel);
  u.h[1] = *(const v8h*)(rowbase + 16 + sel);
  return u.v;
}

__device__ inline void zero_v8f(v8f& x) {
  #pragma unroll
  for (int i = 0; i < 8; i++) x[i] = 0.0f;
}

// ---------------------------------------------------------------------------
// fp32 -> f16 weight conversion (grid-stride)
// ---------------------------------------------------------------------------
__global__ void cvt_f16_kernel(const float* __restrict__ src,
                               half_t* __restrict__ dst, int n) {
  for (int i = blockIdx.x * blockDim.x + threadIdx.x; i < n;
       i += gridDim.x * blockDim.x)
    dst[i] = (half_t)src[i];
}

// ---------------------------------------------------------------------------
// LayerNorm (reference semantics: unbiased std, divide by std+eps), f16 out.
// One wave per row of 1024; 8 rows per 256-thread block.
// ---------------------------------------------------------------------------
__global__ __launch_bounds__(256)
void layernorm_kernel(const float* __restrict__ x, const float* __restrict__ g,
                      const float* __restrict__ be, half_t* __restrict__ out) {
  const int lane = threadIdx.x & 31, wave = threadIdx.x >> 5;
  const int row = blockIdx.x * 8 + wave;
  const float* xr = x + (size_t)row * D_MODEL;
  float v[32];
  float s = 0.f, ss = 0.f;
  #pragma unroll
  for (int i = 0; i < 32; i++) {
    v[i] = xr[lane + i * 32];
    s += v[i];
    ss += v[i] * v[i];
  }
  #pragma unroll
  for (int off = 16; off > 0; off >>= 1) {
    s  += __shfl_xor(s,  off, 32);
    ss += __shfl_xor(ss, off, 32);
  }
  float mean = s * (1.0f / D_MODEL);
  float var  = (ss - (float)D_MODEL * mean * mean) * (1.0f / (D_MODEL - 1));
  float inv  = 1.0f / (sqrtf(fmaxf(var, 0.0f)) + LN_EPS);
  half_t* orow = out + (size_t)row * D_MODEL;
  #pragma unroll
  for (int i = 0; i < 32; i++) {
    int c = lane + i * 32;
    orow[c] = (half_t)(g[c] * (v[i] - mean) * inv + be[c]);
  }
}

// ---------------------------------------------------------------------------
// WMMA GEMM: out[M][N] = epilogue( A[M][K](f16) * W[N][K]^T(f16) )
// Block tile 128x128, double-buffered LDS, 8 waves as 4(M) x 2(N); each wave
// 32x64 = 2x4 fragments -> 8 v_wmma per k-step from 6 fragment loads.
// Epilogue: (+bias)*scale, optional ReLU, optional +residual, f16/f32 out.
// ---------------------------------------------------------------------------
__device__ inline void stage_tiles(half_t (*as)[40], half_t (*ws)[40],
                                   const half_t* A, const half_t* W, int tM,
                                   int tN, int k0, int Kdim, int tid) {
  #pragma unroll
  for (int i = 0; i < 2; i++) {
    int chunk = tid + i * 256;               // 0..511 : 128 rows x 4 chunks
    int row = chunk >> 2, ko = (chunk & 3) * 8;
    cp16_g2l(&A[(size_t)(tM + row) * Kdim + k0 + ko], &as[row][ko]);
    cp16_g2l(&W[(size_t)(tN + row) * Kdim + k0 + ko], &ws[row][ko]);
  }
}

template <bool OUT_F16, bool RELU, bool ADD_RES>
__global__ __launch_bounds__(256)
void gemm_f16_kernel(const half_t* __restrict__ A,     // [M][K]
                     const half_t* __restrict__ W,     // [N][K]
                     const float* __restrict__ bias,   // [N]
                     const float* __restrict__ res,    // [M][N] or null
                     void* __restrict__ outp, int Mdim, int Ndim, int Kdim,
                     float out_scale) {
  __shared__ __align__(16) half_t As[2][128][40];
  __shared__ __align__(16) half_t Ws[2][128][40];
  const int tid = threadIdx.x;
  const int lane = tid & 31, wave = tid >> 5;
  const int wm = wave & 3, wn = wave >> 2;
  const int tM = blockIdx.y * 128, tN = blockIdx.x * 128;
  const int sel = (lane < 16) ? 0 : 8;
  const int ln15 = lane & 15;

  v8f acc[2][4];
  #pragma unroll
  for (int i = 0; i < 2; i++)
    #pragma unroll
    for (int j = 0; j < 4; j++) zero_v8f(acc[i][j]);

  // Prologue: stage tile 0.
  stage_tiles(As[0], Ws[0], A, W, tM, tN, 0, Kdim, tid);
  async_fence();
  __syncthreads();

  const int nk = Kdim >> 5;
  for (int ki = 0; ki < nk; ki++) {
    const int buf = ki & 1;
    // Issue async copies for the next tile; overlaps with the WMMAs below.
    if (ki + 1 < nk)
      stage_tiles(As[buf ^ 1], Ws[buf ^ 1], A, W, tM, tN, (ki + 1) * 32, Kdim,
                  tid);

    v16h a0 = frag_ld(&As[buf][wm * 32 + 0 + ln15][0], sel);
    v16h a1 = frag_ld(&As[buf][wm * 32 + 16 + ln15][0], sel);
    #pragma unroll
    for (int fn = 0; fn < 4; fn++) {
      v16h b = frag_ld(&Ws[buf][wn * 64 + fn * 16 + ln15][0], sel);
      acc[0][fn] = wmma16(a0, b, acc[0][fn]);
      acc[1][fn] = wmma16(a1, b, acc[1][fn]);
    }
    async_fence();   // next tile landed
    __syncthreads(); // everyone done reading buf before it is overwritten
  }

  const int mh = (lane < 16) ? 0 : 8;  // C layout: lanes 16-31 hold rows M+8
  #pragma unroll
  for (int fm = 0; fm < 2; fm++) {
    #pragma unroll
    for (int fn = 0; fn < 4; fn++) {
      int n = tN + wn * 64 + fn * 16 + ln15;
      float bv = bias[n];
      #pragma unroll
      for (int r = 0; r < 8; r++) {
        int m = tM + wm * 32 + fm * 16 + r + mh;
        float val = (acc[fm][fn][r] + bv) * out_scale;
        if (RELU) val = val > 0.0f ? val : 0.0f;
        if (ADD_RES) val += res[(size_t)m * Ndim + n];
        if (OUT_F16)
          ((half_t*)outp)[(size_t)m * Ndim + n] = (half_t)val;
        else
          ((float*)outp)[(size_t)m * Ndim + n] = val;
      }
    }
  }
}

// ---------------------------------------------------------------------------
// Fused attention: one block per (64-query tile, head, batch); 4 waves,
// each wave owns 16 query rows. Two-pass softmax (stats, then PV) so no
// online rescale of the accumulator is needed; scores never touch HBM.
// Q is pre-scaled by 1/sqrt(dk) in the Q-projection epilogue.
// ---------------------------------------------------------------------------
__device__ inline void stage_qk64(half_t (*dst)[72], const half_t* src,
                                  size_t base, int tid) {
  // 64 rows x 64 f16 = 512 16B chunks; 128 threads x 4.
  #pragma unroll
  for (int i = 0; i < 4; i++) {
    int chunk = tid + i * 128;
    int row = chunk >> 3, ko = (chunk & 7) * 8;
    cp16_g2l(&src[base + (size_t)row * D_MODEL + ko], &dst[row][ko]);
  }
}

__global__ __launch_bounds__(128)
void attention_kernel(const half_t* __restrict__ q, const half_t* __restrict__ k,
                      const half_t* __restrict__ v, const int* __restrict__ mask,
                      half_t* __restrict__ ctx) {
  __shared__ __align__(16) half_t Qs[64][72];
  __shared__ __align__(16) half_t Ks[64][72];
  __shared__ __align__(16) half_t Vts[64][72];     // transposed: [d][key]
  __shared__ __align__(16) half_t Ps[4][16][72];   // per-wave exp(S-m) tile

  const int tid = threadIdx.x, lane = tid & 31, wave = tid >> 5;
  const int qt = blockIdx.x, h = blockIdx.y, b = blockIdx.z;
  const int sel = (lane < 16) ? 0 : 8;
  const int ln15 = lane & 15;
  const int mh = (lane < 16) ? 0 : 8;
  const size_t bh = (size_t)b * SEQ * D_MODEL + (size_t)h * D_K;
  const int* mrowp = mask + (size_t)b * SEQ;

  stage_qk64(Qs, q, bh + (size_t)(qt * 64) * D_MODEL, tid);

  float mrow[8], lrow[8];
  #pragma unroll
  for (int r = 0; r < 8; r++) { mrow[r] = -1e30f; lrow[r] = 0.0f; }

  // ---- pass A: row max / sum of exp ----
  for (int j = 0; j < SEQ / 64; j++) {
    __syncthreads();
    stage_qk64(Ks, k, bh + (size_t)(j * 64) * D_MODEL, tid);
    async_fence();
    __syncthreads();
    int mk[4];
    v8f c[4];
    #pragma unroll
    for (int n0 = 0; n0 < 4; n0++) {
      mk[n0] = mrowp[j * 64 + n0 * 16 + ln15];
      zero_v8f(c[n0]);
    }
    #pragma unroll
    for (int kk = 0; kk < 64; kk += 32) {
      v16h a = frag_ld(&Qs[wave * 16 + ln15][kk], sel);
      #pragma unroll
      for (int n0 = 0; n0 < 4; n0++) {
        v16h bf = frag_ld(&Ks[n0 * 16 + ln15][kk], sel);
        c[n0] = wmma16(a, bf, c[n0]);
      }
    }
    #pragma unroll
    for (int r = 0; r < 8; r++) {
      float mx = -1e30f;
      #pragma unroll
      for (int n0 = 0; n0 < 4; n0++)
        mx = fmaxf(mx, mk[n0] ? c[n0][r] : -1e9f);
      #pragma unroll
      for (int off = 1; off < 16; off <<= 1)
        mx = fmaxf(mx, __shfl_xor(mx, off, 32));
      float mn = fmaxf(mrow[r], mx);
      float se = 0.0f;
      #pragma unroll
      for (int n0 = 0; n0 < 4; n0++)
        se += __expf((mk[n0] ? c[n0][r] : -1e9f) - mn);
      #pragma unroll
      for (int off = 1; off < 16; off <<= 1) se += __shfl_xor(se, off, 32);
      lrow[r] = lrow[r] * __expf(mrow[r] - mn) + se;
      mrow[r] = mn;
    }
  }

  // ---- pass B: acc += exp(S - m) * V ----
  v8f acc[4];
  #pragma unroll
  for (int nb = 0; nb < 4; nb++) zero_v8f(acc[nb]);

  for (int j = 0; j < SEQ / 64; j++) {
    __syncthreads();
    stage_qk64(Ks, k, bh + (size_t)(j * 64) * D_MODEL, tid);
    for (int i = tid; i < 64 * 64; i += 128) {   // V transpose: scalar staging
      int r = i >> 6, d = i & 63;
      Vts[d][r] = v[bh + (size_t)(j * 64 + r) * D_MODEL + d];
    }
    async_fence();
    __syncthreads();
    int mk[4];
    v8f c[4];
    #pragma unroll
    for (int n0 = 0; n0 < 4; n0++) {
      mk[n0] = mrowp[j * 64 + n0 * 16 + ln15];
      zero_v8f(c[n0]);
    }
    #pragma unroll
    for (int kk = 0; kk < 64; kk += 32) {
      v16h a = frag_ld(&Qs[wave * 16 + ln15][kk], sel);
      #pragma unroll
      for (int n0 = 0; n0 < 4; n0++) {
        v16h bf = frag_ld(&Ks[n0 * 16 + ln15][kk], sel);
        c[n0] = wmma16(a, bf, c[n0]);
      }
    }
    // P = exp(S - m) into this wave's LDS tile (C layout -> row-major)
    #pragma unroll
    for (int n0 = 0; n0 < 4; n0++)
      #pragma unroll
      for (int r = 0; r < 8; r++) {
        float s = mk[n0] ? c[n0][r] : -1e9f;
        Ps[wave][r + mh][n0 * 16 + ln15] = (half_t)__expf(s - mrow[r]);
      }
    __syncthreads();
    #pragma unroll
    for (int kk = 0; kk < 64; kk += 32) {
      v16h a = frag_ld(&Ps[wave][ln15][kk], sel);
      #pragma unroll
      for (int nb = 0; nb < 4; nb++) {
        v16h bf = frag_ld(&Vts[nb * 16 + ln15][kk], sel);
        acc[nb] = wmma16(a, bf, acc[nb]);
      }
    }
  }

  #pragma unroll
  for (int nb = 0; nb < 4; nb++)
    #pragma unroll
    for (int r = 0; r < 8; r++) {
      int qrow = qt * 64 + wave * 16 + r + mh;
      int d = nb * 16 + ln15;
      ctx[bh + (size_t)qrow * D_MODEL + d] = (half_t)(acc[nb][r] / lrow[r]);
    }
}

// ---------------------------------------------------------------------------
extern "C" void kernel_launch(void* const* d_in, const int* in_sizes, int n_in,
                              void* d_out, int out_size, void* d_ws,
                              size_t ws_size, hipStream_t stream) {
  const float* x    = (const float*)d_in[0];
  const int*   mask = (const int*)d_in[1];
  const float* wq = (const float*)d_in[2];  const float* bq = (const float*)d_in[3];
  const float* wk = (const float*)d_in[4];  const float* bk = (const float*)d_in[5];
  const float* wv = (const float*)d_in[6];  const float* bv = (const float*)d_in[7];
  const float* wo = (const float*)d_in[8];  const float* bo = (const float*)d_in[9];
  const float* w1 = (const float*)d_in[10]; const float* b1 = (const float*)d_in[11];
  const float* w2 = (const float*)d_in[12]; const float* b2 = (const float*)d_in[13];
  const float* g1 = (const float*)d_in[14]; const float* be1 = (const float*)d_in[15];
  const float* g2 = (const float*)d_in[16]; const float* be2 = (const float*)d_in[17];

  // Carve workspace (~216 MB total).
  char* p = (char*)d_ws;
  auto carve = [&](size_t bytes) {
    void* r = (void*)p;
    p += (bytes + 255) & ~(size_t)255;
    return r;
  };
  half_t* wq16 = (half_t*)carve((size_t)D_MODEL * D_MODEL * 2);
  half_t* wk16 = (half_t*)carve((size_t)D_MODEL * D_MODEL * 2);
  half_t* wv16 = (half_t*)carve((size_t)D_MODEL * D_MODEL * 2);
  half_t* wo16 = (half_t*)carve((size_t)D_MODEL * D_MODEL * 2);
  half_t* w116 = (half_t*)carve((size_t)D_FF * D_MODEL * 2);
  half_t* w216 = (half_t*)carve((size_t)D_MODEL * D_FF * 2);
  half_t* h16  = (half_t*)carve((size_t)M_TOT * D_MODEL * 2);
  half_t* q16  = (half_t*)carve((size_t)M_TOT * D_MODEL * 2);
  half_t* k16  = (half_t*)carve((size_t)M_TOT * D_MODEL * 2);
  half_t* v16  = (half_t*)carve((size_t)M_TOT * D_MODEL * 2);
  half_t* ctx16 = (half_t*)carve((size_t)M_TOT * D_MODEL * 2);
  float*  x1   = (float*)carve((size_t)M_TOT * D_MODEL * 4);
  half_t* h216 = (half_t*)carve((size_t)M_TOT * D_MODEL * 2);
  half_t* ff1  = (half_t*)carve((size_t)M_TOT * D_FF * 2);

  // Weight conversion (f32 -> f16, layout preserved [N][K]).
  cvt_f16_kernel<<<256, 256, 0, stream>>>(wq, wq16, D_MODEL * D_MODEL);
  cvt_f16_kernel<<<256, 256, 0, stream>>>(wk, wk16, D_MODEL * D_MODEL);
  cvt_f16_kernel<<<256, 256, 0, stream>>>(wv, wv16, D_MODEL * D_MODEL);
  cvt_f16_kernel<<<256, 256, 0, stream>>>(wo, wo16, D_MODEL * D_MODEL);
  cvt_f16_kernel<<<512, 256, 0, stream>>>(w1, w116, D_FF * D_MODEL);
  cvt_f16_kernel<<<512, 256, 0, stream>>>(w2, w216, D_MODEL * D_FF);

  dim3 blk(256);
  dim3 gN1024(D_MODEL / 128, M_TOT / 128);
  dim3 gN4096(D_FF / 128, M_TOT / 128);

  // h = LN1(x)
  layernorm_kernel<<<M_TOT / 8, blk, 0, stream>>>(x, g1, be1, h16);
  // Q (scaled by 1/sqrt(dk)), K, V projections
  gemm_f16_kernel<true, false, false><<<gN1024, blk, 0, stream>>>(
      h16, wq16, bq, nullptr, q16, M_TOT, D_MODEL, D_MODEL, 0.125f);
  gemm_f16_kernel<true, false, false><<<gN1024, blk, 0, stream>>>(
      h16, wk16, bk, nullptr, k16, M_TOT, D_MODEL, D_MODEL, 1.0f);
  gemm_f16_kernel<true, false, false><<<gN1024, blk, 0, stream>>>(
      h16, wv16, bv, nullptr, v16, M_TOT, D_MODEL, D_MODEL, 1.0f);
  // attention -> ctx
  attention_kernel<<<dim3(SEQ / 64, HEADS, BATCH), dim3(128), 0, stream>>>(
      q16, k16, v16, mask, ctx16);
  // x1 = x + ctx @ wo^T + bo
  gemm_f16_kernel<false, false, true><<<gN1024, blk, 0, stream>>>(
      ctx16, wo16, bo, x, x1, M_TOT, D_MODEL, D_MODEL, 1.0f);
  // h2 = LN2(x1)
  layernorm_kernel<<<M_TOT / 8, blk, 0, stream>>>(x1, g2, be2, h216);
  // ff1 = relu(h2 @ w1^T + b1)
  gemm_f16_kernel<true, true, false><<<gN4096, blk, 0, stream>>>(
      h216, w116, b1, nullptr, ff1, M_TOT, D_FF, D_MODEL, 1.0f);
  // out = x1 + ff1 @ w2^T + b2
  gemm_f16_kernel<false, false, true><<<gN1024, blk, 0, stream>>>(
      ff1, w216, b2, x1, (float*)d_out, M_TOT, D_MODEL, D_FF, 1.0f);
}
